// SAGE_RNN_4209067950697
// MI455X (gfx1250) — compile-verified, matrix operand order checked by reading, not run on hardware
//
#include <hip/hip_runtime.h>
#include <hip/hip_bf16.h>

typedef __bf16 bf16;
typedef __attribute__((ext_vector_type(16))) __bf16 v16bf;
typedef __attribute__((ext_vector_type(8)))  __bf16 v8bf;
typedef __attribute__((ext_vector_type(8)))  float  v8f;

#define N0 32768
#define N1 8192
#define N2 4096
#define TSTEPS 6
#define FDIM 431
#define FAN 8
#define HID 64

// ---------------------------------------------------------------------------
// WMMA tile loader: matrix R is (rows x K) row-major bf16, ld == K.
// Loads the 16x32 operand tile at (row0, k0) into the wave's A/B operand.
// Layout per CDNA5 ISA 7.12.2 (16-bit A 16x32): lane m = l&15, k-half = l>>4;
// VGPR0..3 = K (k0 + kh*8 + 0..7), VGPR4..7 = K (k0 + 16 + kh*8 + 0..7).
// Weights stored (N x K) row-major use the same loader for the B operand.
// ---------------------------------------------------------------------------
__device__ __forceinline__ v16bf load_tile(const bf16* __restrict__ base,
                                           int row0, int K, int k0, int lane) {
  int r  = row0 + (lane & 15);
  int kh = (lane >> 4) << 3;
  const bf16* p = base + (size_t)r * K + k0 + kh;
  v8bf lo = *(const v8bf*)(p);
  v8bf hi = *(const v8bf*)(p + 16);
  v16bf out;
#pragma unroll
  for (int i = 0; i < 8; ++i) { out[i] = lo[i]; out[i + 8] = hi[i]; }
  return out;
}

// ---------------------------------------------------------------------------
// Dual-accumulation GEMM: C(M x N) = A1(M x K1) @ B1(N x K1)^T
//                                  + A2(M x K2) @ B2(N x K2)^T + bias [, relu]
// One 16x16 tile per wave, 4 waves per block. Grids exact -> EXEC all ones.
// ---------------------------------------------------------------------------
template <int K1, int K2>
__global__ __launch_bounds__(128) void gemm2_kernel(
    const bf16* __restrict__ A1, const bf16* __restrict__ B1,
    const bf16* __restrict__ A2, const bf16* __restrict__ B2,
    const float* __restrict__ bias1, const float* __restrict__ bias2,
    float* __restrict__ Cf, bf16* __restrict__ Cb, int ldc,
    int ntiles, int do_relu) {
  int lane = threadIdx.x & 31;
  int wave = threadIdx.x >> 5;
  int tile = blockIdx.x * 4 + wave;
  int mt = tile / ntiles, nt = tile % ntiles;
  int m0 = mt << 4, n0 = nt << 4;

  v8f c = {};
#pragma unroll
  for (int k0 = 0; k0 < K1; k0 += 32) {
    v16bf a = load_tile(A1, m0, K1, k0, lane);
    v16bf b = load_tile(B1, n0, K1, k0, lane);
    c = __builtin_amdgcn_wmma_f32_16x16x32_bf16(false, a, false, b,
                                                (short)0, c, false, false);
  }
#pragma unroll
  for (int k0 = 0; k0 < K2; k0 += 32) {
    v16bf a = load_tile(A2, m0, K2, k0, lane);
    v16bf b = load_tile(B2, n0, K2, k0, lane);
    c = __builtin_amdgcn_wmma_f32_16x16x32_bf16(false, a, false, b,
                                                (short)0, c, false, false);
  }

  // C layout: lane n = n0 + (l&15); VGPR r holds row m0 + r + 8*(l>=16)
  int n = n0 + (lane & 15);
  float bsum = bias1[n] + (bias2 ? bias2[n] : 0.f);
#pragma unroll
  for (int r = 0; r < 8; ++r) {
    int m = m0 + r + ((lane >> 4) << 3);
    float v = c[r] + bsum;
    if (do_relu) v = fmaxf(v, 0.f);
    if (Cf) Cf[(size_t)m * ldc + n] = v;
    if (Cb) Cb[(size_t)m * ldc + n] = (bf16)v;
  }
}

// ---------------------------------------------------------------------------
// CNN: wave-cooperative, one node per wave, LDS staging. Produces h (99-dim)
// padded to 128 cols in bf16 for the SAGE WMMA GEMMs.
// ---------------------------------------------------------------------------
struct CnnParams {
  const float *x;
  const float *c1w, *c1b, *c2w, *c2b, *c3w, *c3b, *c4w, *c4b, *wfW, *wfb;
  const float *s1w, *s1b, *s2w, *s2b, *s3w, *s3b, *sfW, *sfb;
  bf16* hout;  // N0 x 128
  int t;
};

#define CNN_WPB 2
__global__ __launch_bounds__(64) void cnn_kernel(CnnParams p) {
  __shared__ float sm[CNN_WPB][3264];
  int wave = threadIdx.x >> 5;
  int lane = threadIdx.x & 31;
  int node = blockIdx.x * CNN_WPB + wave;

  float* xin  = sm[wave];         // 431 (round 432)
  float* bufA = xin  + 432;       // 6*8*22  = 1056
  float* bufB = bufA + 1056;      // 6*12*10 = 720
  float* bufC = bufB + 720;       // 6*16*4  = 384
  float* bufD = bufC + 384;       // 6*20    = 120
  float* bufE = bufD + 120;       // 10*4*4  = 160
  float* bufF = bufE + 160;       // 10*8*2  = 160
  float* bufG = bufF + 160;       // 10*12   = 120
  float* bufH = bufG + 120;       // 80

  const float* xrow = p.x + ((size_t)node * TSTEPS + p.t) * FDIM;
  for (int i = lane; i < FDIM; i += 32) xin[i] = xrow[i];
  __syncthreads();

  // --- w branch: conv1(1->8,k9)+relu+pool2 : len 52 -> 44 -> 22
  for (int task = lane; task < 6 * 8 * 22; task += 32) {
    int pp = task % 22, ch = (task / 22) & 7, seq = task / (22 * 8);
    const float* in = xin + seq * 52;
    const float* w = p.c1w + ch * 9;
    float b = p.c1b[ch], acc = 0.f;
#pragma unroll
    for (int u = 0; u < 2; ++u) {
      float yv = b; int pos = 2 * pp + u;
#pragma unroll
      for (int k = 0; k < 9; ++k) yv += in[pos + k] * w[k];
      acc += fmaxf(yv, 0.f);
    }
    bufA[(seq * 8 + ch) * 22 + pp] = acc * 0.5f;
  }
  __syncthreads();
  // conv2(8->12,k3)+relu+pool2 : 22 -> 20 -> 10
  for (int task = lane; task < 6 * 12 * 10; task += 32) {
    int pp = task % 10, ch = (task / 10) % 12, seq = task / 120;
    float b = p.c2b[ch], acc = 0.f;
#pragma unroll
    for (int u = 0; u < 2; ++u) {
      float yv = b; int pos = 2 * pp + u;
      for (int ic = 0; ic < 8; ++ic)
#pragma unroll
        for (int k = 0; k < 3; ++k)
          yv += bufA[(seq * 8 + ic) * 22 + pos + k] * p.c2w[(ch * 8 + ic) * 3 + k];
      acc += fmaxf(yv, 0.f);
    }
    bufB[(seq * 12 + ch) * 10 + pp] = acc * 0.5f;
  }
  __syncthreads();
  // conv3(12->16,k3)+relu+pool2 : 10 -> 8 -> 4
  for (int task = lane; task < 6 * 16 * 4; task += 32) {
    int pp = task & 3, ch = (task >> 2) & 15, seq = task >> 6;
    float b = p.c3b[ch], acc = 0.f;
#pragma unroll
    for (int u = 0; u < 2; ++u) {
      float yv = b; int pos = 2 * pp + u;
      for (int ic = 0; ic < 12; ++ic)
#pragma unroll
        for (int k = 0; k < 3; ++k)
          yv += bufB[(seq * 12 + ic) * 10 + pos + k] * p.c3w[(ch * 12 + ic) * 3 + k];
      acc += fmaxf(yv, 0.f);
    }
    bufC[(seq * 16 + ch) * 4 + pp] = acc * 0.5f;
  }
  __syncthreads();
  // conv4(16->20,k3)+relu+pool2 : 4 -> 2 -> 1
  for (int task = lane; task < 6 * 20; task += 32) {
    int ch = task % 20, seq = task / 20;
    float b = p.c4b[ch], acc = 0.f;
#pragma unroll
    for (int u = 0; u < 2; ++u) {
      float yv = b;
      for (int ic = 0; ic < 16; ++ic)
#pragma unroll
        for (int k = 0; k < 3; ++k)
          yv += bufC[(seq * 16 + ic) * 4 + u + k] * p.c4w[(ch * 16 + ic) * 3 + k];
      acc += fmaxf(yv, 0.f);
    }
    bufD[seq * 20 + ch] = acc * 0.5f;
  }
  __syncthreads();
  // FC w: 120 -> 40, relu
  for (int o = lane; o < 40; o += 32) {
    float acc = p.wfb[o];
    for (int j = 0; j < 120; ++j) acc += bufD[j] * p.wfW[o * 120 + j];
    bufH[o] = fmaxf(acc, 0.f);
  }

  // --- s branch: s1(1->4,k3)+relu+pool2 : 10 -> 8 -> 4
  for (int task = lane; task < 10 * 4 * 4; task += 32) {
    int pp = task & 3, ch = (task >> 2) & 3, seq = task >> 4;
    const float* in = xin + 312 + seq * 10;
    float b = p.s1b[ch], acc = 0.f;
#pragma unroll
    for (int u = 0; u < 2; ++u) {
      float yv = b; int pos = 2 * pp + u;
#pragma unroll
      for (int k = 0; k < 3; ++k) yv += in[pos + k] * p.s1w[ch * 3 + k];
      acc += fmaxf(yv, 0.f);
    }
    bufE[(seq * 4 + ch) * 4 + pp] = acc * 0.5f;
  }
  __syncthreads();
  // s2(4->8,k3)+relu : 4 -> 2
  for (int task = lane; task < 10 * 8 * 2; task += 32) {
    int pos = task & 1, ch = (task >> 1) & 7, seq = task >> 4;
    float yv = p.s2b[ch];
    for (int ic = 0; ic < 4; ++ic)
#pragma unroll
      for (int k = 0; k < 3; ++k)
        yv += bufE[(seq * 4 + ic) * 4 + pos + k] * p.s2w[(ch * 4 + ic) * 3 + k];
    bufF[(seq * 8 + ch) * 2 + pos] = fmaxf(yv, 0.f);
  }
  __syncthreads();
  // s3(8->12,k2)+relu : 2 -> 1
  for (int task = lane; task < 10 * 12; task += 32) {
    int ch = task % 12, seq = task / 12;
    float yv = p.s3b[ch];
    for (int ic = 0; ic < 8; ++ic)
#pragma unroll
      for (int k = 0; k < 2; ++k)
        yv += bufF[(seq * 8 + ic) * 2 + k] * p.s3w[(ch * 8 + ic) * 2 + k];
    bufG[seq * 12 + ch] = fmaxf(yv, 0.f);
  }
  __syncthreads();
  // FC s: 120 -> 40, relu
  for (int o = lane; o < 40; o += 32) {
    float acc = p.sfb[o];
    for (int j = 0; j < 120; ++j) acc += bufG[j] * p.sfW[o * 120 + j];
    bufH[40 + o] = fmaxf(acc, 0.f);
  }
  __syncthreads();

  // concat [hw(40), hs(40), X[:,412:431](19)] padded to 128, bf16
  bf16* orow = p.hout + (size_t)node * 128;
  for (int c = lane; c < 128; c += 32) {
    float v = 0.f;
    if (c < 80)      v = bufH[c];
    else if (c < 99) v = xin[412 + (c - 80)];
    orow[c] = (bf16)v;
  }
}

// ---------------------------------------------------------------------------
// Support kernels
// ---------------------------------------------------------------------------
__global__ void gather_mean_kernel(const bf16* __restrict__ src,
                                   const int* __restrict__ nbr,
                                   bf16* __restrict__ dst, int rows, int cols) {
  int idx = blockIdx.x * blockDim.x + threadIdx.x;
  if (idx >= rows * cols) return;
  int i = idx / cols, c = idx % cols;
  const int* nb = nbr + (size_t)i * FAN;
  float s = 0.f;
#pragma unroll
  for (int j = 0; j < FAN; ++j) s += (float)src[(size_t)nb[j] * cols + c];
  dst[idx] = (bf16)(s * 0.125f);
}

__global__ void convert_pad_kernel(const float* __restrict__ in,
                                   bf16* __restrict__ out, int N, int Kin, int Kpad) {
  int idx = blockIdx.x * blockDim.x + threadIdx.x;
  if (idx >= N * Kpad) return;
  int r = idx / Kpad, k = idx % Kpad;
  out[idx] = (bf16)((k < Kin) ? in[(size_t)r * Kin + k] : 0.f);
}

__global__ void zero_bf16_kernel(bf16* p, int n) {
  int i = blockIdx.x * blockDim.x + threadIdx.x;
  if (i < n) p[i] = (bf16)0.f;
}
__global__ void zero_f32_kernel(float* p, int n) {
  int i = blockIdx.x * blockDim.x + threadIdx.x;
  if (i < n) p[i] = 0.f;
}

// fill hs columns 64..95: col 64 = y_pad[:,t], 65..95 = 0 (K padding)
__global__ void fill_hs_kernel(bf16* __restrict__ hs, const float* __restrict__ y, int t) {
  int idx = blockIdx.x * blockDim.x + threadIdx.x;
  if (idx >= N2 * 32) return;
  int i = idx >> 5, c = 64 + (idx & 31);
  float v = 0.f;
  if (c == 64 && t > 0) v = y[(size_t)i * (TSTEPS - 1) + (t - 1)];
  hs[((size_t)t * N2 + i) * 96 + c] = (bf16)v;
}

__global__ void lstm_elem_kernel(const float* __restrict__ G,
                                 float* __restrict__ h, float* __restrict__ cst,
                                 bf16* __restrict__ hbf) {
  int idx = blockIdx.x * blockDim.x + threadIdx.x;
  if (idx >= N2 * HID) return;
  int col = idx % HID;
  const float* g = G + (size_t)(idx / HID) * 256;
  float gi = g[col], gf = g[64 + col], gg = g[128 + col], go = g[192 + col];
  float si = 1.f / (1.f + __expf(-gi));
  float sf = 1.f / (1.f + __expf(-gf));
  float so = 1.f / (1.f + __expf(-go));
  float cc = sf * cst[idx] + si * tanhf(gg);
  float hh = so * tanhf(cc);
  cst[idx] = cc; h[idx] = hh; hbf[idx] = (bf16)hh;
}

__global__ void head_kernel(const float* __restrict__ h,
                            const float* __restrict__ r1W, const float* __restrict__ r1b,
                            const float* __restrict__ r2W, const float* __restrict__ r2b,
                            float* __restrict__ out) {
  int i = blockIdx.x * blockDim.x + threadIdx.x;
  if (i >= N2) return;
  const float* hr = h + (size_t)i * HID;
  float acc = r2b[0];
  for (int o = 0; o < 32; ++o) {
    float t = r1b[o];
#pragma unroll
    for (int j = 0; j < 64; ++j) t += hr[j] * r1W[o * 64 + j];
    acc += fmaxf(t, 0.f) * r2W[o];
  }
  out[i] = acc;
}

// ---------------------------------------------------------------------------
// Reference dict order:
//  0:x 1:y 2:nbr0 3:nbr1 4:c1w 5:c1b 6:c2w 7:c2b 8:c3w 9:c3b 10:c4w 11:c4b
// 12:wfW 13:wfb 14:s1w 15:s1b 16:s2w 17:s2b 18:s3w 19:s3b 20:sfW 21:sfb
// 22:W0s 23:W0n 24:b0 25:W1s 26:W1n 27:b1 28:Wih 29:Whh 30:bih 31:bhh
// 32:r1W 33:r1b 34:r2W 35:r2b
// ---------------------------------------------------------------------------
extern "C" void kernel_launch(void* const* d_in, const int* in_sizes, int n_in,
                              void* d_out, int out_size, void* d_ws, size_t ws_size,
                              hipStream_t stream) {
  const float* x   = (const float*)d_in[0];
  const float* y   = (const float*)d_in[1];
  const int* nbr0  = (const int*)d_in[2];
  const int* nbr1  = (const int*)d_in[3];
  const float *c1w = (const float*)d_in[4],  *c1b = (const float*)d_in[5];
  const float *c2w = (const float*)d_in[6],  *c2b = (const float*)d_in[7];
  const float *c3w = (const float*)d_in[8],  *c3b = (const float*)d_in[9];
  const float *c4w = (const float*)d_in[10], *c4b = (const float*)d_in[11];
  const float *wfW = (const float*)d_in[12], *wfb = (const float*)d_in[13];
  const float *s1w = (const float*)d_in[14], *s1b = (const float*)d_in[15];
  const float *s2w = (const float*)d_in[16], *s2b = (const float*)d_in[17];
  const float *s3w = (const float*)d_in[18], *s3b = (const float*)d_in[19];
  const float *sfW = (const float*)d_in[20], *sfb = (const float*)d_in[21];
  const float *W0s = (const float*)d_in[22], *W0n = (const float*)d_in[23];
  const float *b0  = (const float*)d_in[24];
  const float *W1s = (const float*)d_in[25], *W1n = (const float*)d_in[26];
  const float *b1  = (const float*)d_in[27];
  const float *Wih = (const float*)d_in[28], *Whh = (const float*)d_in[29];
  const float *bih = (const float*)d_in[30], *bhh = (const float*)d_in[31];
  const float *r1W = (const float*)d_in[32], *r1b = (const float*)d_in[33];
  const float *r2W = (const float*)d_in[34], *r2b = (const float*)d_in[35];

  char* ws = (char*)d_ws;
  size_t off = 0;
  auto alloc = [&](size_t bytes) -> void* {
    void* p = ws + off;
    off = (off + bytes + 255) & ~(size_t)255;
    return p;
  };
  bf16* h_all  = (bf16*)alloc((size_t)N0 * 128 * 2);
  bf16* hn0    = (bf16*)alloc((size_t)N1 * 128 * 2);
  bf16* h1     = (bf16*)alloc((size_t)N1 * 64 * 2);
  bf16* hn1    = (bf16*)alloc((size_t)N2 * 64 * 2);
  bf16* hs     = (bf16*)alloc((size_t)TSTEPS * N2 * 96 * 2);
  bf16* W0s_b  = (bf16*)alloc(64 * 128 * 2);
  bf16* W0n_b  = (bf16*)alloc(64 * 128 * 2);
  bf16* W1s_b  = (bf16*)alloc(64 * 64 * 2);
  bf16* W1n_b  = (bf16*)alloc(64 * 64 * 2);
  bf16* Wih_b  = (bf16*)alloc(256 * 96 * 2);
  bf16* Whh_b  = (bf16*)alloc(256 * 64 * 2);
  float* G     = (float*)alloc((size_t)N2 * 256 * 4);
  float* hstate= (float*)alloc((size_t)N2 * HID * 4);
  float* cstate= (float*)alloc((size_t)N2 * HID * 4);
  bf16* hp     = (bf16*)alloc((size_t)N2 * HID * 2);

  const int th = 256;
  // weight conversion (fp32 -> bf16, K padded to multiple of 32)
  convert_pad_kernel<<<(64 * 128 + th - 1) / th, th, 0, stream>>>(W0s, W0s_b, 64, 99, 128);
  convert_pad_kernel<<<(64 * 128 + th - 1) / th, th, 0, stream>>>(W0n, W0n_b, 64, 99, 128);
  convert_pad_kernel<<<(64 * 64 + th - 1) / th, th, 0, stream>>>(W1s, W1s_b, 64, 64, 64);
  convert_pad_kernel<<<(64 * 64 + th - 1) / th, th, 0, stream>>>(W1n, W1n_b, 64, 64, 64);
  convert_pad_kernel<<<(256 * 96 + th - 1) / th, th, 0, stream>>>(Wih, Wih_b, 256, 65, 96);
  convert_pad_kernel<<<(256 * 64 + th - 1) / th, th, 0, stream>>>(Whh, Whh_b, 256, 64, 64);
  // LSTM state init
  zero_f32_kernel<<<(N2 * HID + th - 1) / th, th, 0, stream>>>(cstate, N2 * HID);
  zero_bf16_kernel<<<(N2 * HID + th - 1) / th, th, 0, stream>>>(hp, N2 * HID);

  for (int t = 0; t < TSTEPS; ++t) {
    CnnParams cp{x, c1w, c1b, c2w, c2b, c3w, c3b, c4w, c4b, wfW, wfb,
                 s1w, s1b, s2w, s2b, s3w, s3b, sfW, sfb, h_all, t};
    cnn_kernel<<<N0 / CNN_WPB, 32 * CNN_WPB, 0, stream>>>(cp);

    gather_mean_kernel<<<(N1 * 128 + th - 1) / th, th, 0, stream>>>(h_all, nbr0, hn0, N1, 128);
    // SAGE0: M=8192, N=64, K=128+128 -> h1 (bf16)
    gemm2_kernel<128, 128><<<(N1 / 16) * 4 / 4, 128, 0, stream>>>(
        h_all, W0s_b, hn0, W0n_b, b0, nullptr, nullptr, h1, 64, 4, 1);
    gather_mean_kernel<<<(N2 * 64 + th - 1) / th, th, 0, stream>>>(h1, nbr1, hn1, N2, 64);
    // SAGE1: M=4096, N=64, K=64+64 -> hs[t] cols 0..63 (bf16, ld 96)
    gemm2_kernel<64, 64><<<(N2 / 16) * 4 / 4, 128, 0, stream>>>(
        h1, W1s_b, hn1, W1n_b, b1, nullptr,
        nullptr, hs + (size_t)t * N2 * 96, 96, 4, 1);
    fill_hs_kernel<<<(N2 * 32 + th - 1) / th, th, 0, stream>>>(hs, y, t);
  }

  // LSTM: per step G = hs[t] @ Wih^T + hp @ Whh^T + bih + bhh, then gates
  for (int t = 0; t < TSTEPS; ++t) {
    gemm2_kernel<96, 64><<<(N2 / 16) * 16 / 4, 128, 0, stream>>>(
        hs + (size_t)t * N2 * 96, Wih_b, hp, Whh_b, bih, bhh,
        G, nullptr, 256, 16, 0);
    lstm_elem_kernel<<<(N2 * HID + th - 1) / th, th, 0, stream>>>(G, hstate, cstate, hp);
  }

  head_kernel<<<(N2 + th - 1) / th, th, 0, stream>>>(hstate, r1W, r1b, r2W, r2b, (float*)d_out);
  (void)in_sizes; (void)n_in; (void)out_size; (void)ws_size;
}
